// SelectiveSSM_33741263077639
// MI455X (gfx1250) — matrix operand order repaired and checked by reading, not compiled, not run on hardware
//
#include <hip/hip_runtime.h>
#include <stdint.h>

#define Dq     1024
#define STATEq 16
#define CONVq  4
#define Bq     8
#define Lq     2048
#define Mq     (Bq * Lq)   // 16384

typedef __attribute__((ext_vector_type(16))) __bf16 v16bf;
typedef __attribute__((ext_vector_type(8)))  float  v8f;

__device__ __forceinline__ uint16_t f2bf(float f) {
    uint32_t u = __builtin_bit_cast(uint32_t, f);
    u += 0x7FFFu + ((u >> 16) & 1u);          // round-to-nearest-even
    return (uint16_t)(u >> 16);
}

// ---------------------------------------------------------------------------
// CDNA5 async global->LDS (ASYNCcnt) with __has_builtin fallback
// ---------------------------------------------------------------------------
#if __has_builtin(__builtin_amdgcn_global_load_async_to_lds_b128)
#define HAVE_ASYNC_LDS 1
typedef int vs4i __attribute__((vector_size(16)));   // matches builtin's param type
__device__ __forceinline__ void async_b128(const void* g, void* l) {
    __builtin_amdgcn_global_load_async_to_lds_b128((vs4i*)g, (vs4i*)l, 0, 0);
}
#if __has_builtin(__builtin_amdgcn_s_wait_asynccnt)
#define WAIT_ASYNC(n) __builtin_amdgcn_s_wait_asynccnt(n)
#else
#define WAIT_ASYNC(n) asm volatile("s_wait_asynccnt %0" ::"n"(n) : "memory")
#endif
#else
#define HAVE_ASYNC_LDS 0
#endif

// ---------------------------------------------------------------------------
// f32 -> bf16 buffer conversion (weights)
// ---------------------------------------------------------------------------
__global__ __launch_bounds__(256) void f2bf_kernel(const float* __restrict__ in,
                                                   uint16_t* __restrict__ out, int n) {
    int i = blockIdx.x * 256 + threadIdx.x;
    if (i < n) out[i] = f2bf(in[i]);
}

// ---------------------------------------------------------------------------
// causal depthwise conv(4) + SiLU; also emits bf16 copies of xc and x
// ---------------------------------------------------------------------------
__global__ __launch_bounds__(256) void conv_silu_kernel(
    const float* __restrict__ x, const float* __restrict__ cw,
    const float* __restrict__ cb, float* __restrict__ xc,
    uint16_t* __restrict__ xcb, uint16_t* __restrict__ xb) {
    size_t idx = (size_t)blockIdx.x * 256 + threadIdx.x;   // over B*L*D
    int d  = (int)(idx % Dq);
    size_t bl = idx / Dq;
    int l  = (int)(bl % Lq);
    float acc = cb[d];
    const float* cwd = cw + d * CONVq;
#pragma unroll
    for (int j = 0; j < CONVq; ++j) {
        int ls = l + j - (CONVq - 1);
        if (ls >= 0) acc += x[(bl + (size_t)(j - (CONVq - 1))) * Dq + d] * cwd[j];
    }
    float s = acc / (1.f + __expf(-acc));   // SiLU
    xc[idx]  = s;
    xcb[idx] = f2bf(s);
    xb[idx]  = f2bf(x[idx]);
}

// ---------------------------------------------------------------------------
// WMMA bf16 GEMM: out[m,n] = epi( sum_k A[m,k] * W[n,k] + bias[n] )
//   EPI 0: softplus -> f32      EPI 1: sigmoid * aux -> bf16     EPI 2: +bias -> f32
// 128x128 block tile, BK=64, 8 waves (2x4), 64x32 per wave, 16 WMMA per K-tile.
// Double-buffered LDS filled by async global->LDS (memory never touches VGPRs).
// ---------------------------------------------------------------------------
#define TM 128
#define TN 128
#define TK 64
#define LDSA (TK + 8)   // 72 elements -> 144B row stride (conflict-free b128)

template <int EPI>
__global__ __launch_bounds__(256) void gemm_bf16(
    const uint16_t* __restrict__ A, const uint16_t* __restrict__ Wt,
    const float* __restrict__ bias, const float* __restrict__ aux,
    float* __restrict__ outf, uint16_t* __restrict__ outb) {
    constexpr int GK = 1024, GN = 1024;
    constexpr int NT = GK / TK;
    __shared__ uint16_t lA[2][TM * LDSA];
    __shared__ uint16_t lB[2][TN * LDSA];
    const int tid  = threadIdx.x;
    const int m0   = blockIdx.x * TM;
    const int n0   = blockIdx.y * TN;
    const int wid  = tid >> 5, lane = tid & 31;
    const int wm   = wid >> 2, wn = wid & 3;   // 2 (M) x 4 (N) waves
    const int l16  = lane & 15, lh = lane >> 4;
    const int r0   = tid >> 3;                 // staging row base
    const int c8   = (tid & 7) * 8;            // staging col (elements)

    v8f acc[4][2] = {};

#if HAVE_ASYNC_LDS
    // issue 8 async b128 fills (4 A rows + 4 W rows) for one K-tile
    auto issue = [&](int kt, int buf) {
        const int k0 = kt * TK;
#pragma unroll
        for (int i = 0; i < 4; ++i) {
            int row = r0 + i * 32;
            async_b128(A  + (size_t)(m0 + row) * GK + k0 + c8, &lA[buf][row * LDSA + c8]);
            async_b128(Wt + (size_t)(n0 + row) * GK + k0 + c8, &lB[buf][row * LDSA + c8]);
        }
    };
    issue(0, 0);
#endif

    for (int kt = 0; kt < NT; ++kt) {
        const int cur = kt & 1;
#if HAVE_ASYNC_LDS
        if (kt + 1 < NT) { issue(kt + 1, cur ^ 1); WAIT_ASYNC(8); }
        else             { WAIT_ASYNC(0); }
        __syncthreads();
#else
        {
            const int k0 = kt * TK;
            uint4 ra[4], rb[4];
#pragma unroll
            for (int i = 0; i < 4; ++i) {
                int row = r0 + i * 32;
                ra[i] = *(const uint4*)(A  + (size_t)(m0 + row) * GK + k0 + c8);
                rb[i] = *(const uint4*)(Wt + (size_t)(n0 + row) * GK + k0 + c8);
            }
            __syncthreads();
#pragma unroll
            for (int i = 0; i < 4; ++i) {
                int row = r0 + i * 32;
                *(uint4*)&lA[cur][row * LDSA + c8] = ra[i];
                *(uint4*)&lB[cur][row * LDSA + c8] = rb[i];
            }
            __syncthreads();
        }
#endif
#pragma unroll
        for (int kk = 0; kk < TK; kk += 32) {
            union { uint4 u[2]; v16bf v; } af[4], bw[2];
#pragma unroll
            for (int i = 0; i < 4; ++i) {   // A frag: row=lane%16, K {0-7,16-23}/{8-15,24-31}
                int r  = wm * 64 + i * 16 + l16;
                int ka = kk + lh * 8;
                af[i].u[0] = *(const uint4*)&lA[cur][r * LDSA + ka];
                af[i].u[1] = *(const uint4*)&lA[cur][r * LDSA + ka + 16];
            }
#pragma unroll
            for (int j = 0; j < 2; ++j) {   // B frag: col=lane%16, K {0-7,16-23}/{8-15,24-31}
                int r  = wn * 32 + j * 16 + l16;
                int kb = kk + lh * 8;
                bw[j].u[0] = *(const uint4*)&lB[cur][r * LDSA + kb];
                bw[j].u[1] = *(const uint4*)&lB[cur][r * LDSA + kb + 16];
            }
#pragma unroll
            for (int i = 0; i < 4; ++i)
#pragma unroll
                for (int j = 0; j < 2; ++j)
                    acc[i][j] = __builtin_amdgcn_wmma_f32_16x16x32_bf16(
                        false, af[i].v, false, bw[j].v, (short)0, acc[i][j], false, false);
        }
        __syncthreads();
    }

#pragma unroll
    for (int j = 0; j < 2; ++j) {
        int n = n0 + wn * 32 + j * 16 + l16;
        float bv = bias[n];
#pragma unroll
        for (int i = 0; i < 4; ++i) {
#pragma unroll
            for (int v = 0; v < 8; ++v) {
                int m = m0 + wm * 64 + i * 16 + lh * 8 + v;   // C/D: vgpr v -> M=v (+8 hi lanes)
                size_t idx = (size_t)m * GN + n;
                float r = acc[i][j][v] + bv;
                if (EPI == 0) {
                    outf[idx] = (r > 20.f) ? r : log1pf(__expf(r));
                } else if (EPI == 1) {
                    float g = 1.f / (1.f + __expf(-r));
                    outb[idx] = f2bf(g * aux[idx]);
                } else {
                    outf[idx] = r;
                }
            }
        }
    }
}

// ---------------------------------------------------------------------------
// B/C projections: Bm = xc @ Wb^T, Cm = xc @ Wc^T  (N=16 each, f32)
// block = 8 m-rows staged in LDS x 32 (n): n<16 -> B, n>=16 -> C
// ---------------------------------------------------------------------------
__global__ __launch_bounds__(256) void bc_kernel(
    const float* __restrict__ xc, const float* __restrict__ Wb,
    const float* __restrict__ Wc, float* __restrict__ Bm, float* __restrict__ Cm) {
    __shared__ float lx[8 * Dq];
    int tid = threadIdx.x;
    size_t mbase = (size_t)blockIdx.x * 8;
    const float4* src = (const float4*)(xc + mbase * Dq);
    float4* dst = (float4*)lx;
#pragma unroll
    for (int i = 0; i < 8; ++i) dst[tid + i * 256] = src[tid + i * 256];
    __syncthreads();
    int n = tid & 31, ro = tid >> 5;
    const float* w  = (n < 16) ? (Wb + n * Dq) : (Wc + (n - 16) * Dq);
    const float* xr = lx + ro * Dq;
    float acc = 0.f;
    for (int k = 0; k < Dq; k += 4) {
        float4 wv = *(const float4*)(w + k);
        acc += xr[k] * wv.x + xr[k + 1] * wv.y + xr[k + 2] * wv.z + xr[k + 3] * wv.w;
    }
    size_t m = mbase + ro;
    if (n < 16) Bm[m * 16 + n] = acc;
    else        Cm[m * 16 + (n - 16)] = acc;
}

// ---------------------------------------------------------------------------
// sequential selective scan: thread = (b, d), 16 states in registers,
// B/C timestep values broadcast across the wave via __shfl (wave32)
// ---------------------------------------------------------------------------
__global__ __launch_bounds__(256) void scan_kernel(
    const float* __restrict__ delta, const float* __restrict__ xc,
    const float* __restrict__ Bm, const float* __restrict__ Cm,
    const float* __restrict__ log_a, float* __restrict__ y) {
    const int b    = blockIdx.y;
    const int d    = blockIdx.x * 256 + threadIdx.x;
    const int lane = threadIdx.x & 31;
    float Ad[STATEq], h[STATEq];
#pragma unroll
    for (int n = 0; n < STATEq; ++n) { Ad[n] = -__expf(log_a[d * STATEq + n]); h[n] = 0.f; }
    const float* Bb = Bm + (size_t)b * Lq * STATEq;
    const float* Cb = Cm + (size_t)b * Lq * STATEq;
    size_t base = ((size_t)b * Lq) * Dq + d;
    for (int l = 0; l < Lq; ++l) {
        float bc = (lane < 16) ? Bb[l * 16 + lane] : Cb[l * 16 + lane - 16];
        float dt = delta[base];
        float xv = xc[base];
        float dx = dt * xv;
        float yv = 0.f;
#pragma unroll
        for (int n = 0; n < STATEq; ++n) {
            float bn = __shfl(bc, n, 32);
            float cn = __shfl(bc, 16 + n, 32);
            h[n] = __expf(Ad[n] * dt) * h[n] + dx * bn;
            yv += h[n] * cn;
        }
        y[base] = yv;
        base += Dq;
    }
}

// ---------------------------------------------------------------------------
extern "C" void kernel_launch(void* const* d_in, const int* in_sizes, int n_in,
                              void* d_out, int out_size, void* d_ws, size_t ws_size,
                              hipStream_t stream) {
    (void)in_sizes; (void)n_in; (void)out_size; (void)ws_size;
    const float* x      = (const float*)d_in[0];
    const float* conv_w = (const float*)d_in[1];
    const float* conv_b = (const float*)d_in[2];
    const float* Wd     = (const float*)d_in[3];
    const float* bd     = (const float*)d_in[4];
    const float* Wb     = (const float*)d_in[5];
    const float* Wc     = (const float*)d_in[6];
    const float* log_a  = (const float*)d_in[7];
    const float* Wg     = (const float*)d_in[8];
    const float* bg     = (const float*)d_in[9];
    const float* Wo     = (const float*)d_in[10];
    const float* bo     = (const float*)d_in[11];

    char* ws = (char*)d_ws;
    size_t off = 0;
    auto alloc = [&](size_t bytes) -> void* {
        void* p = ws + off;
        off = (off + bytes + 255) & ~(size_t)255;
        return p;
    };
    const size_t MD = (size_t)Mq * Dq;
    float*    xc_f  = (float*)   alloc(MD * 4);
    uint16_t* xc_b  = (uint16_t*)alloc(MD * 2);
    uint16_t* x_b   = (uint16_t*)alloc(MD * 2);
    float*    delta = (float*)   alloc(MD * 4);
    float*    Bm    = (float*)   alloc((size_t)Mq * STATEq * 4);
    float*    Cm    = (float*)   alloc((size_t)Mq * STATEq * 4);
    float*    yb    = (float*)   alloc(MD * 4);
    uint16_t* Wd_b  = (uint16_t*)alloc((size_t)Dq * Dq * 2);
    uint16_t* Wg_b  = (uint16_t*)alloc((size_t)Dq * Dq * 2);
    uint16_t* Wo_b  = (uint16_t*)alloc((size_t)Dq * Dq * 2);
    uint16_t* t_b   = xc_b;   // reuse: xc_bf16 consumed by the delta GEMM

    const int nW = Dq * Dq;
    f2bf_kernel<<<nW / 256, 256, 0, stream>>>(Wd, Wd_b, nW);
    f2bf_kernel<<<nW / 256, 256, 0, stream>>>(Wg, Wg_b, nW);
    f2bf_kernel<<<nW / 256, 256, 0, stream>>>(Wo, Wo_b, nW);

    conv_silu_kernel<<<(unsigned)(MD / 256), 256, 0, stream>>>(x, conv_w, conv_b, xc_f, xc_b, x_b);

    dim3 ggrid(Mq / TM, Dq / TN);
    gemm_bf16<0><<<ggrid, 256, 0, stream>>>(xc_b, Wd_b, bd, nullptr, delta, nullptr);

    bc_kernel<<<Mq / 8, 256, 0, stream>>>(xc_f, Wb, Wc, Bm, Cm);

    scan_kernel<<<dim3(Dq / 256, Bq), 256, 0, stream>>>(delta, xc_f, Bm, Cm, log_a, yb);

    gemm_bf16<1><<<ggrid, 256, 0, stream>>>(x_b, Wg_b, bg, yb, nullptr, t_b);

    gemm_bf16<2><<<ggrid, 256, 0, stream>>>(t_b, Wo_b, bo, nullptr, (float*)d_out, nullptr);
}
